// Rea_15135464751210
// MI455X (gfx1250) — matrix operand order repaired and checked by reading, not compiled, not run on hardware
//
#include <hip/hip_runtime.h>
#include <hip/hip_bf16.h>

// ---------------------------------------------------------------------------
// MI455X (gfx1250) implementation.
// Pipeline (all bf16 WMMA GEMMs, f32 accumulate):
//   xb  = bf16(x);  Wb* = bf16(W*);  xbT = bf16(x)^T per batch
//   qb  = xb @ Wb1^T + b1          (bf16)
//   kb  = xb @ Wb2^T + b2          (bf16)
//   att = (qb @ kb^T) * (1/28)     (bf16, 64MB -> L2 resident)
//   t1  = att @ xbT^T              (bf16, reuses xb buffer)
//   out = t1 @ Wb3^T + b3          (f32 = g_x1)
//   g1T = bf16(out)^T per batch    (reuses qb buffer)
//   att = masked_softmax(att,mask) (in place, bf16)
//   t2  = att @ g1T^T              (bf16, reuses kb buffer)
//   out += t2 @ Wb4^T + b4         (f32 accumulate epilogue)
// GEMM: 128x128x32 tiles, 8 waves, double-buffered LDS, async global->LDS
// staging (GLOBAL_LOAD_ASYNC_TO_LDS_B128 + s_wait_asynccnt) when available.
// Workspace need: ~173 MB.
// ---------------------------------------------------------------------------

typedef __attribute__((ext_vector_type(16))) __bf16 v16bf;
typedef __attribute__((ext_vector_type(8)))  float  v8f;
typedef __attribute__((ext_vector_type(4)))  int    v4i;

#define BM 128
#define BN 128
#define BK 32
#define LDSP 96   // LDS row pitch in bytes (32 bf16 = 64B data + 32B pad, keeps 16B align)

#if defined(__has_builtin)
#if __has_builtin(__builtin_amdgcn_global_load_async_to_lds_b128)
#define HAS_ASYNC_LDS 1
#endif
#endif

union FragU { uint4 u4[2]; v16bf v; };

__device__ __forceinline__ __bf16 f2bf(float f) { return (__bf16)f; }

// Copy 16 bytes global -> LDS. Async (ASYNCcnt-tracked) on gfx1250 toolchains
// that expose the builtin; plain b128 copy otherwise.
// Builtin signature (from clang diagnostic): arg0 = v4i* in AS1 (global),
// arg1 = LDS destination, then imm offset + imm cpol.
__device__ __forceinline__ void stage128(const unsigned char* g, unsigned char* l)
{
#if HAS_ASYNC_LDS
    __builtin_amdgcn_global_load_async_to_lds_b128(
        (__attribute__((address_space(1))) v4i*)g,
        (__attribute__((address_space(3))) v4i*)l,
        0, 0);
#else
    *(uint4*)l = *(const uint4*)g;
#endif
}

#if HAS_ASYNC_LDS
#if __has_builtin(__builtin_amdgcn_s_wait_asynccnt)
#define WAIT_ASYNC(n) __builtin_amdgcn_s_wait_asynccnt(n)
#else
#define WAIT_ASYNC(n) asm volatile("s_wait_asynccnt %0" :: "i"(n) : "memory")
#endif
#else
#define WAIT_ASYNC(n) ((void)0)
#endif

// C[M,N] = scale * (A[M,K] @ Bt[N,K]^T) (+ bias[N]) (+= C if ACCUM)
// A, Bt row-major bf16. OUT_BF16 ? C is bf16 : C is f32.
template<bool OUT_BF16, bool BIAS, bool ACCUM>
__global__ __launch_bounds__(256)
void gemm_bf16(const __bf16* __restrict__ A, const __bf16* __restrict__ Bt,
               void* __restrict__ C, const float* __restrict__ bias,
               float scale, int K, int ldA, int ldB, int ldC,
               long long batchStrideA, long long batchStrideB, long long batchStrideC)
{
    __shared__ __align__(16) unsigned char shA[2][BM * LDSP];
    __shared__ __align__(16) unsigned char shB[2][BN * LDSP];

    const int tid  = threadIdx.x;
    const int z    = blockIdx.z;
    const int m0   = blockIdx.y * BM;
    const int n0   = blockIdx.x * BN;

    const int wave  = tid >> 5;
    const int lane  = tid & 31;
    const int wm    = wave & 3;    // 4 waves along M -> 32 rows each
    const int wn    = wave >> 2;   // 2 waves along N -> 64 cols each
    const int laneN = lane & 15;
    const int hsel  = lane >> 4;   // 0: lanes 0-15, 1: lanes 16-31

    const unsigned char* Ab = (const unsigned char*)(A  + (size_t)z * batchStrideA);
    const unsigned char* Bb = (const unsigned char*)(Bt + (size_t)z * batchStrideB);

    // Straight-line staging assignment: thread -> chunks {tid, tid+256} of the
    // 512 16B chunks per 128x32 tile (row = chunk>>2, 16B col = chunk&3).
    const int srow = tid >> 2;            // 0..63
    const int scol = (tid & 3) << 4;      // 0,16,32,48 bytes within 64B row
    const unsigned char* gA0 = Ab + 2ull * (size_t)(m0 + srow)      * ldA + scol;
    const unsigned char* gA1 = Ab + 2ull * (size_t)(m0 + srow + 64) * ldA + scol;
    const unsigned char* gB0 = Bb + 2ull * (size_t)(n0 + srow)      * ldB + scol;
    const unsigned char* gB1 = Bb + 2ull * (size_t)(n0 + srow + 64) * ldB + scol;
    const unsigned lA0 = srow * LDSP + scol;
    const unsigned lA1 = (srow + 64) * LDSP + scol;

#define STAGE_TILE(buf, kk) do {                                   \
        const size_t kb_ = 2ull * (size_t)(kk);                    \
        stage128(gA0 + kb_, shA[buf] + lA0);                       \
        stage128(gA1 + kb_, shA[buf] + lA1);                       \
        stage128(gB0 + kb_, shB[buf] + lA0);                       \
        stage128(gB1 + kb_, shB[buf] + lA1);                       \
    } while (0)

    v8f acc[2][4] = {};

    STAGE_TILE(0, 0);                      // prologue: tile 0 -> buffer 0
    int cur = 0;

    for (int k0 = 0; k0 < K; k0 += BK) {
        const bool hasNext = (k0 + BK) < K;
        if (hasNext) STAGE_TILE(cur ^ 1, k0 + BK);   // overlap copy of tile i+1

        // Async loads retire in order: <=4 outstanding => current tile landed.
        if (hasNext) WAIT_ASYNC(4); else WAIT_ASYNC(0);
        __syncthreads();

        // ---- fragment loads per ISA 16-bit A(16x32)/B(32x16) layouts ----
        const unsigned char* sA = shA[cur];
        const unsigned char* sB = shB[cur];
        FragU a[2], b[4];
        #pragma unroll
        for (int i = 0; i < 2; ++i) {
            const int r = wm * 32 + i * 16 + laneN;
            const int off = hsel * 16;               // lane>=16 -> K 8..15 / 24..31
            a[i].u4[0] = *(const uint4*)(sA + r * LDSP + off);
            a[i].u4[1] = *(const uint4*)(sA + r * LDSP + 32 + off);
        }
        #pragma unroll
        for (int j = 0; j < 4; ++j) {
            const int n = wn * 64 + j * 16 + laneN;
            const int off = hsel * 16;               // lane>=16 -> K 8..15 / 24..31
            b[j].u4[0] = *(const uint4*)(sB + n * LDSP + off);
            b[j].u4[1] = *(const uint4*)(sB + n * LDSP + off + 32);
        }

        #pragma unroll
        for (int i = 0; i < 2; ++i)
            #pragma unroll
            for (int j = 0; j < 4; ++j)
                acc[i][j] = __builtin_amdgcn_wmma_f32_16x16x32_bf16(
                    false, a[i].v, false, b[j].v, (short)0, acc[i][j], false, false);

        __syncthreads();                   // all readers of buf `cur` done
        cur ^= 1;
    }
#undef STAGE_TILE

    // ---- epilogue: C layout VGPR r -> row (r + hsel*8), col laneN ----
    const size_t cBatch = (size_t)z * batchStrideC;
    #pragma unroll
    for (int j = 0; j < 4; ++j) {
        const int col = n0 + wn * 64 + j * 16 + laneN;
        const float bv = BIAS ? bias[col] : 0.0f;
        #pragma unroll
        for (int i = 0; i < 2; ++i) {
            const int mBase = m0 + wm * 32 + i * 16 + hsel * 8;
            const v8f cv = acc[i][j];
            #pragma unroll
            for (int r = 0; r < 8; ++r) {
                const float v = cv[r] * scale + bv;
                const size_t idx = cBatch + (size_t)(mBase + r) * ldC + col;
                if (OUT_BF16) {
                    ((__bf16*)C)[idx] = f2bf(v);
                } else {
                    float* p = (float*)C + idx;
                    *p = ACCUM ? (*p + v) : v;
                }
            }
        }
    }
}

// out[cols,rows] = bf16(in[rows,cols]^T), batched via blockIdx.z
__global__ __launch_bounds__(256)
void transpose_cvt(const float* __restrict__ in, __bf16* __restrict__ out,
                   int rows, int cols, long long inStride, long long outStride)
{
    __shared__ float tile[32][33];
    const int z  = blockIdx.z;
    const int c0 = blockIdx.x * 32;
    const int r0 = blockIdx.y * 32;
    const int tx = threadIdx.x & 31;
    const int ty = threadIdx.x >> 5;   // 0..7
    const float* ib = in  + (size_t)z * inStride;
    __bf16*      ob = out + (size_t)z * outStride;
    #pragma unroll
    for (int k = 0; k < 32; k += 8)
        tile[ty + k][tx] = ib[(size_t)(r0 + ty + k) * cols + (c0 + tx)];
    __syncthreads();
    #pragma unroll
    for (int k = 0; k < 32; k += 8)
        ob[(size_t)(c0 + ty + k) * rows + (r0 + tx)] = f2bf(tile[tx][ty + k]);
}

__global__ __launch_bounds__(256)
void cvt_f32_bf16(const float* __restrict__ in, __bf16* __restrict__ out, size_t n)
{
    const size_t i = (size_t)blockIdx.x * 256 + threadIdx.x;
    if (i < n) out[i] = f2bf(in[i]);
}

// In-place masked softmax over rows of att (bf16), one 2048-wide row per block.
__global__ __launch_bounds__(256)
void masked_softmax_inplace(__bf16* __restrict__ att, const int* __restrict__ mask, int N)
{
    __shared__ float red[256];
    const int tid = threadIdx.x;
    const size_t base = ((size_t)blockIdx.y * N + blockIdx.x) * (size_t)N;
    __bf16* arow = att + base;
    const int* mrow = mask + base;

    float v[8];
    float lmax = -3.4e38f;
    #pragma unroll
    for (int e = 0; e < 8; ++e) {
        const int idx = tid + e * 256;
        float a = (float)arow[idx];
        if (mrow[idx] == 1) a = -1e9f;
        v[e] = a;
        lmax = fmaxf(lmax, a);
    }
    red[tid] = lmax; __syncthreads();
    for (int s = 128; s > 0; s >>= 1) {
        if (tid < s) red[tid] = fmaxf(red[tid], red[tid + s]);
        __syncthreads();
    }
    const float m = red[0]; __syncthreads();

    float lsum = 0.0f;
    #pragma unroll
    for (int e = 0; e < 8; ++e) { v[e] = __expf(v[e] - m); lsum += v[e]; }
    red[tid] = lsum; __syncthreads();
    for (int s = 128; s > 0; s >>= 1) {
        if (tid < s) red[tid] += red[tid + s];
        __syncthreads();
    }
    const float inv = 1.0f / red[0];
    #pragma unroll
    for (int e = 0; e < 8; ++e) arow[tid + e * 256] = f2bf(v[e] * inv);
}

extern "C" void kernel_launch(void* const* d_in, const int* in_sizes, int n_in,
                              void* d_out, int out_size, void* d_ws, size_t ws_size,
                              hipStream_t stream)
{
    const int B = 8, N = 2048, D = 768;
    typedef long long ll;

    const float* x    = (const float*)d_in[0];
    const int*   mask = (const int*)  d_in[1];
    const float* W1   = (const float*)d_in[2];
    const float* W2   = (const float*)d_in[3];
    const float* W3   = (const float*)d_in[4];
    const float* W4   = (const float*)d_in[5];
    const float* b1   = (const float*)d_in[6];
    const float* b2   = (const float*)d_in[7];
    const float* b3   = (const float*)d_in[8];
    const float* b4   = (const float*)d_in[9];
    float* out = (float*)d_out;

    unsigned char* ws = (unsigned char*)d_ws;
    size_t off = 0;
    auto alloc = [&](size_t bytes) -> void* {
        void* p = ws + off;
        off += (bytes + 255) & ~(size_t)255;
        return p;
    };
    __bf16* Wb1 = (__bf16*)alloc((size_t)D * D * 2);
    __bf16* Wb2 = (__bf16*)alloc((size_t)D * D * 2);
    __bf16* Wb3 = (__bf16*)alloc((size_t)D * D * 2);
    __bf16* Wb4 = (__bf16*)alloc((size_t)D * D * 2);
    __bf16* xb  = (__bf16*)alloc((size_t)B * N * D * 2);   // later reused as t1
    __bf16* xbT = (__bf16*)alloc((size_t)B * D * N * 2);
    __bf16* qb  = (__bf16*)alloc((size_t)B * N * D * 2);   // later reused as g_x1^T
    __bf16* kb  = (__bf16*)alloc((size_t)B * N * D * 2);   // later reused as t2
    __bf16* att = (__bf16*)alloc((size_t)B * N * N * 2);   // scores, then softmax in place

    // 1) precision conversions
    cvt_f32_bf16<<<(B * N * D) / 256, 256, 0, stream>>>(x, xb, (size_t)B * N * D);
    cvt_f32_bf16<<<(D * D) / 256, 256, 0, stream>>>(W1, Wb1, (size_t)D * D);
    cvt_f32_bf16<<<(D * D) / 256, 256, 0, stream>>>(W2, Wb2, (size_t)D * D);
    cvt_f32_bf16<<<(D * D) / 256, 256, 0, stream>>>(W3, Wb3, (size_t)D * D);
    cvt_f32_bf16<<<(D * D) / 256, 256, 0, stream>>>(W4, Wb4, (size_t)D * D);
    transpose_cvt<<<dim3(D / 32, N / 32, B), 256, 0, stream>>>(
        x, xbT, N, D, (ll)N * D, (ll)D * N);

    // 2) q/k projections: [B*N,768] x [768,768]
    gemm_bf16<true, true, false><<<dim3(D / 128, (B * N) / 128, 1), 256, 0, stream>>>(
        xb, Wb1, qb, b1, 1.0f, D, D, D, D, 0, 0, 0);
    gemm_bf16<true, true, false><<<dim3(D / 128, (B * N) / 128, 1), 256, 0, stream>>>(
        xb, Wb2, kb, b2, 1.0f, D, D, D, D, 0, 0, 0);

    // 3) att = q @ k^T / 28 per batch
    gemm_bf16<true, false, false><<<dim3(N / 128, N / 128, B), 256, 0, stream>>>(
        qb, kb, att, nullptr, 1.0f / 28.0f, D, D, D, N,
        (ll)N * D, (ll)N * D, (ll)N * N);

    // 4) t1 = att @ x  (B-operand = x^T, rows contiguous) -> reuse xb buffer
    gemm_bf16<true, false, false><<<dim3(D / 128, N / 128, B), 256, 0, stream>>>(
        att, xbT, xb, nullptr, 1.0f, N, N, N, D,
        (ll)N * N, (ll)D * N, (ll)N * D);

    // 5) g_x1 = t1 @ W3^T + b3  -> f32 straight into d_out
    gemm_bf16<false, true, false><<<dim3(D / 128, (B * N) / 128, 1), 256, 0, stream>>>(
        xb, Wb3, out, b3, 1.0f, D, D, D, D, 0, 0, 0);

    // 6) g_x1^T in bf16 (reuse qb buffer) for the second aggregation GEMM
    transpose_cvt<<<dim3(D / 32, N / 32, B), 256, 0, stream>>>(
        out, qb, N, D, (ll)N * D, (ll)D * N);

    // 7) masked softmax in place on att
    masked_softmax_inplace<<<dim3(N, B), 256, 0, stream>>>(att, mask, N);

    // 8) t2 = softmax(att) @ g_x1  -> reuse kb buffer
    gemm_bf16<true, false, false><<<dim3(D / 128, N / 128, B), 256, 0, stream>>>(
        att, qb, kb, nullptr, 1.0f, N, N, N, D,
        (ll)N * N, (ll)D * N, (ll)N * D);

    // 9) out += t2 @ W4^T + b4
    gemm_bf16<false, true, true><<<dim3(D / 128, (B * N) / 128, 1), 256, 0, stream>>>(
        kb, Wb4, out, b4, 1.0f, D, D, D, D, 0, 0, 0);
}
